// StarTransformerLayer_15187004359089
// MI455X (gfx1250) — matrix operand order repaired and checked by reading, not compiled, not run on hardware
//
#include <hip/hip_runtime.h>

// ---------------------------------------------------------------------------
// Star-Transformer layer for MI455X (gfx1250, wave32, WMMA).
// All heavy GEMMs (6 per cycle + 2 one-time) run as bf16 WMMA 16x16x32 with
// f32 accumulate; full 256x256 bf16 weight (128 KB) staged in LDS per WG via
// the Tensor Data Mover (TDM) when available, else a cooperative copy.
// ---------------------------------------------------------------------------

#define DMODEL 256
#define NHEADS 8
#define DHEAD  32
#define SEQ    4096
#define BATCH  8
#define NTOK   (BATCH * SEQ)      // 32768 tokens
#define LN_EPS 1e-12f
#define ATTN_SCALE 0.17677669529663687f  // 1/sqrt(32)

typedef __attribute__((ext_vector_type(16))) __bf16 v16bf;
typedef __attribute__((ext_vector_type(8)))  __bf16 v8bf;
typedef __attribute__((ext_vector_type(8)))  float  v8f;
typedef __attribute__((ext_vector_type(4)))  unsigned int u32x4;
typedef __attribute__((ext_vector_type(8)))  int i32x8;
typedef __attribute__((ext_vector_type(4)))  int i32x4;

// ---------------------------------------------------------------- casts ----
__global__ __launch_bounds__(256) void cast_f2b_kernel(
    const float* __restrict__ x, __bf16* __restrict__ y, int n) {
  int i = (blockIdx.x * 256 + threadIdx.x) * 4;
  if (i + 3 < n) {
    float4 v = *(const float4*)(x + i);
    y[i + 0] = (__bf16)v.x;
    y[i + 1] = (__bf16)v.y;
    y[i + 2] = (__bf16)v.z;
    y[i + 3] = (__bf16)v.w;
  }
}

// -------------------------------------------------- s0 = mean_l e[b,l,:] ---
__global__ __launch_bounds__(256) void mean_seq_kernel(
    const float* __restrict__ e, float* __restrict__ s) {
  const int b = blockIdx.x, d = threadIdx.x;
  const float* p = e + (size_t)b * SEQ * DMODEL + d;
  float acc = 0.f;
  for (int l = 0; l < SEQ; ++l) acc += p[(size_t)l * DMODEL];
  s[b * DMODEL + d] = acc * (1.0f / SEQ);
}

// ------------------------------------------------- WMMA GEMM: Y = X@W^T+b --
// X: (NTOK,256) bf16 row-major.  W: (256,256) bf16, W[n][k] (torch Linear).
// Block = 256 thr = 8 waves; wave w computes rows [blk*128+16w, +16) x all 256
// cols as 16 WMMA C-tiles.  Whole W lives in LDS (128 KB dynamic), staged by
// the Tensor Data Mover (TENSOR_LOAD_TO_LDS, TENSORcnt) from wave 0.
// A-fragment (16-bit A 16x32 layout): lanes0-15 row M=lane hold K0..7,K16..23;
// lanes16-31 row M=lane-16 hold K8..15,K24..31  -> two global_load_b128/lane.
// B-fragment: lane half holds K8h..8h+7,K16+8h..+7 of column N=nt*16+(lane&15) from LDS;
// B loads are software-pipelined one tile ahead of the WMMA consuming them.
__global__ __launch_bounds__(256) void gemm_xwt_kernel(
    const __bf16* __restrict__ X, const __bf16* __restrict__ W,
    const float* __restrict__ bias,
    float* __restrict__ Yf, __bf16* __restrict__ Ybf) {
  extern __shared__ __bf16 wlds[];  // 256*256 bf16 = 131072 B, LDS offset 0

#if __has_builtin(__builtin_amdgcn_tensor_load_to_lds)
  if (threadIdx.x < 32) {  // wave 0 issues the DMA (EXEC ignored by TDM)
    const unsigned long long ga = (unsigned long long)(uintptr_t)W;
    // D# group 0: count=1 | lds_addr=0 | global_addr | type=2
    u32x4 g0 = {1u, 0u, (unsigned)(ga & 0xffffffffu),
                (unsigned)((ga >> 32) & 0x01ffffffu) | (2u << 30)};
    // D# group 1: data_size=1 (2B); tensor_dim0=256 @bits[79:48];
    // tensor_dim1=256 @[111:80]; tile_dim0=256 @[127:112]; tile_dim1=256
    // @[143:128]; tensor_dim0_stride=256 @[207:160].
    i32x8 g1 = {0x00010000, (int)(256u << 16), (int)(256u << 16),
                (int)(256u << 16), 0x00000100, 0x00000100, 0, 0};
    i32x4 gz = {0, 0, 0, 0};
#if __clang_major__ >= 23
    i32x8 gz8 = {0, 0, 0, 0, 0, 0, 0, 0};
    __builtin_amdgcn_tensor_load_to_lds(g0, g1, gz, gz, gz8, 0);
#else
    __builtin_amdgcn_tensor_load_to_lds(g0, g1, gz, gz, 0);
#endif
    __builtin_amdgcn_s_wait_tensorcnt(0);
  }
#else
  for (int i = threadIdx.x; i < (DMODEL * DMODEL) / 8; i += 256)
    *(v8bf*)(wlds + i * 8) = *(const v8bf*)(W + i * 8);
#endif
  __syncthreads();

  const int tid = threadIdx.x;
  const int wave = tid >> 5, lane = tid & 31;
  const int half = lane >> 4, lq = lane & 15;
  const int row0 = blockIdx.x * 128 + wave * 16;
  const __bf16* aptr = X + (size_t)(row0 + lq) * DMODEL;

  v8f acc[16] = {};

  for (int kb = 0; kb < 8; ++kb) {
    v8bf a0 = *(const v8bf*)(aptr + kb * 32 + half * 8);
    v8bf a1 = *(const v8bf*)(aptr + kb * 32 + 16 + half * 8);
    v16bf afrag = __builtin_shufflevector(a0, a1, 0, 1, 2, 3, 4, 5, 6, 7, 8, 9,
                                          10, 11, 12, 13, 14, 15);
    if (kb < 7) __builtin_prefetch(aptr + (kb + 1) * 32, 0, 3);
    // double-buffered B-fragment pipeline
    const __bf16* bbase = wlds + (size_t)lq * DMODEL + kb * 32 + half * 8;
    v8bf b0 = *(const v8bf*)bbase;
    v8bf b1 = *(const v8bf*)(bbase + 16);
#pragma unroll
    for (int nt = 0; nt < 16; ++nt) {
      v8bf nb0 = b0, nb1 = b1;
      if (nt < 15) {
        const __bf16* bp = bbase + (size_t)(nt + 1) * 16 * DMODEL;
        nb0 = *(const v8bf*)bp;
        nb1 = *(const v8bf*)(bp + 16);
      }
      v16bf bfrag = __builtin_shufflevector(b0, b1, 0, 1, 2, 3, 4, 5, 6, 7, 8,
                                            9, 10, 11, 12, 13, 14, 15);
      acc[nt] = __builtin_amdgcn_wmma_f32_16x16x32_bf16(
          false, afrag, false, bfrag, (short)0, acc[nt], false, false);
      b0 = nb0;
      b1 = nb1;
    }
  }

  // C/D layout: VGPR r = (M=r, lanes0-15 N=lane) / (M=r+8, lanes16-31)
#pragma unroll
  for (int nt = 0; nt < 16; ++nt) {
    const int n = nt * 16 + lq;
    const float bv = bias[n];
#pragma unroll
    for (int r = 0; r < 8; ++r) {
      const int grow = row0 + r + half * 8;
      const float v = acc[nt][r] + bv;
      if (Yf) Yf[(size_t)grow * DMODEL + n] = v;
      if (Ybf) Ybf[(size_t)grow * DMODEL + n] = (__bf16)v;
    }
  }
}

// ------------------------- small (B,256)x(256,256)^T matvec (+relu+LN) -----
__global__ __launch_bounds__(256) void matvec256_kernel(
    const float* __restrict__ x, const float* __restrict__ W,
    const float* __restrict__ bias, float* __restrict__ yf,
    __bf16* __restrict__ ybf, float* __restrict__ yf2,
    const float* __restrict__ lnw, const float* __restrict__ lnb,
    int relu_ln) {
  __shared__ float xs[DMODEL];
  __shared__ float red[DMODEL];
  const int b = blockIdx.x, n = threadIdx.x;
  xs[n] = x[b * DMODEL + n];
  __syncthreads();
  const float* wr = W + (size_t)n * DMODEL;
  float acc = bias[n];
#pragma unroll 8
  for (int k = 0; k < DMODEL; ++k) acc = fmaf(xs[k], wr[k], acc);
  if (relu_ln) {
    acc = fmaxf(acc, 0.f);
    red[n] = acc;
    __syncthreads();
    for (int s = 128; s > 0; s >>= 1) {
      if (n < s) red[n] += red[n + s];
      __syncthreads();
    }
    const float u = red[0] * (1.f / DMODEL);
    __syncthreads();
    const float d = acc - u;
    red[n] = d * d;
    __syncthreads();
    for (int s = 128; s > 0; s >>= 1) {
      if (n < s) red[n] += red[n + s];
      __syncthreads();
    }
    const float var = red[0] * (1.f / DMODEL);
    acc = lnw[n] * d * rsqrtf(var + LN_EPS) + lnb[n];
  }
  if (yf) yf[b * DMODEL + n] = acc;
  if (ybf) ybf[b * DMODEL + n] = (__bf16)acc;
  if (yf2) yf2[b * DMODEL + n] = acc;
}

// ------------------------------------ per-row relu + LayerNorm (32768 rows) -
__global__ __launch_bounds__(256) void relu_ln_rows_kernel(
    const __bf16* __restrict__ X, const float* __restrict__ lnw,
    const float* __restrict__ lnb, float* __restrict__ hout,
    __bf16* __restrict__ hbf) {
  __shared__ float red[DMODEL];
  const int row = blockIdx.x, n = threadIdx.x;
  float v = (float)X[(size_t)row * DMODEL + n];
  v = fmaxf(v, 0.f);
  red[n] = v;
  __syncthreads();
  for (int s = 128; s > 0; s >>= 1) {
    if (n < s) red[n] += red[n + s];
    __syncthreads();
  }
  const float u = red[0] * (1.f / DMODEL);
  __syncthreads();
  const float d = v - u;
  red[n] = d * d;
  __syncthreads();
  for (int s = 128; s > 0; s >>= 1) {
    if (n < s) red[n] += red[n + s];
    __syncthreads();
  }
  const float var = red[0] * (1.f / DMODEL);
  const float y = lnw[n] * d * rsqrtf(var + LN_EPS) + lnb[n];
  hout[(size_t)row * DMODEL + n] = y;
  hbf[(size_t)row * DMODEL + n] = (__bf16)y;
}

// ------------- satellite attention: per (token, head) softmax over 5 keys ---
// keys order = [h_last, h, h_next, e, s]; h_last[l]=h[l+1], h_next[l]=h[l-1]
__global__ __launch_bounds__(256) void sat_attn_kernel(
    const __bf16* __restrict__ Q, const __bf16* __restrict__ Kh,
    const __bf16* __restrict__ Vh, const __bf16* __restrict__ Ke,
    const __bf16* __restrict__ Ve, const __bf16* __restrict__ Ks,
    const __bf16* __restrict__ Vs, __bf16* __restrict__ ctx) {
  const int gid = blockIdx.x * 256 + threadIdx.x;
  const int h = gid & 7;
  const int t = gid >> 3;  // b*SEQ + l
  const int b = t >> 12, l = t & (SEQ - 1);
  const int lm1 = (l == 0) ? SEQ - 1 : l - 1;
  const int lp1 = (l == SEQ - 1) ? 0 : l + 1;
  const size_t ho = (size_t)h * DHEAD;
  const size_t rb = (size_t)b * SEQ * DMODEL;

  float q[DHEAD];
  {
    const __bf16* qp = Q + (size_t)t * DMODEL + ho;
#pragma unroll
    for (int j = 0; j < DHEAD; ++j) q[j] = (float)qp[j];
  }
  const __bf16* kp[5] = {Kh + rb + (size_t)lp1 * DMODEL + ho,
                         Kh + rb + (size_t)l * DMODEL + ho,
                         Kh + rb + (size_t)lm1 * DMODEL + ho,
                         Ke + (size_t)t * DMODEL + ho,
                         Ks + (size_t)b * DMODEL + ho};
  const __bf16* vp[5] = {Vh + rb + (size_t)lp1 * DMODEL + ho,
                         Vh + rb + (size_t)l * DMODEL + ho,
                         Vh + rb + (size_t)lm1 * DMODEL + ho,
                         Ve + (size_t)t * DMODEL + ho,
                         Vs + (size_t)b * DMODEL + ho};
  float sc[5], m = -1e30f;
#pragma unroll
  for (int i = 0; i < 5; ++i) {
    float d = 0.f;
#pragma unroll
    for (int j = 0; j < DHEAD; ++j) d = fmaf(q[j], (float)kp[i][j], d);
    sc[i] = d * ATTN_SCALE;
    m = fmaxf(m, sc[i]);
  }
  float sum = 0.f;
#pragma unroll
  for (int i = 0; i < 5; ++i) {
    sc[i] = __expf(sc[i] - m);
    sum += sc[i];
  }
  const float inv = 1.f / sum;
  __bf16* cp = ctx + (size_t)t * DMODEL + ho;
#pragma unroll
  for (int j = 0; j < DHEAD; ++j) {
    float a = 0.f;
#pragma unroll
    for (int i = 0; i < 5; ++i) a = fmaf(sc[i] * inv, (float)vp[i][j], a);
    cp[j] = (__bf16)a;
  }
}

// --------- relay attention: (b,head) block; 1 query over 4097 keys ---------
__global__ __launch_bounds__(256) void rel_attn_kernel(
    const float* __restrict__ qs, const __bf16* __restrict__ Kr,
    const __bf16* __restrict__ Vr, const __bf16* __restrict__ k0,
    const __bf16* __restrict__ v0, float* __restrict__ rctx) {
  const int b = blockIdx.x >> 3, h = blockIdx.x & 7;
  const int tid = threadIdx.x;
  __shared__ float qsh[DHEAD];
  __shared__ float red[256];
  __shared__ float part[8][DHEAD];
  if (tid < DHEAD) qsh[tid] = qs[b * DMODEL + h * DHEAD + tid];
  __syncthreads();

  // pass 1: global max of scores
  float lmax = -1e30f;
  for (int i = tid; i < SEQ + 1; i += 256) {
    const __bf16* kp = (i < SEQ)
                           ? (Kr + ((size_t)b * SEQ + i) * DMODEL + h * DHEAD)
                           : (k0 + (size_t)b * DMODEL + h * DHEAD);
    float d = 0.f;
#pragma unroll
    for (int j = 0; j < DHEAD; ++j) d = fmaf(qsh[j], (float)kp[j], d);
    lmax = fmaxf(lmax, d * ATTN_SCALE);
  }
  red[tid] = lmax;
  __syncthreads();
  for (int s = 128; s > 0; s >>= 1) {
    if (tid < s) red[tid] = fmaxf(red[tid], red[tid + s]);
    __syncthreads();
  }
  const float gmax = red[0];
  __syncthreads();

  // pass 2: exp-sum + weighted V accumulation
  float lsum = 0.f;
  float acc[DHEAD];
#pragma unroll
  for (int j = 0; j < DHEAD; ++j) acc[j] = 0.f;
  for (int i = tid; i < SEQ + 1; i += 256) {
    const __bf16 *kp, *vp;
    if (i < SEQ) {
      kp = Kr + ((size_t)b * SEQ + i) * DMODEL + h * DHEAD;
      vp = Vr + ((size_t)b * SEQ + i) * DMODEL + h * DHEAD;
    } else {
      kp = k0 + (size_t)b * DMODEL + h * DHEAD;
      vp = v0 + (size_t)b * DMODEL + h * DHEAD;
    }
    float d = 0.f;
#pragma unroll
    for (int j = 0; j < DHEAD; ++j) d = fmaf(qsh[j], (float)kp[j], d);
    const float e = __expf(d * ATTN_SCALE - gmax);
    lsum += e;
#pragma unroll
    for (int j = 0; j < DHEAD; ++j) acc[j] = fmaf(e, (float)vp[j], acc[j]);
  }
  red[tid] = lsum;
  __syncthreads();
  for (int s = 128; s > 0; s >>= 1) {
    if (tid < s) red[tid] += red[tid + s];
    __syncthreads();
  }
  const float gsum = red[0];

  // reduce acc: wave32 shuffle then cross-wave via LDS
#pragma unroll
  for (int j = 0; j < DHEAD; ++j) {
#pragma unroll
    for (int off = 16; off > 0; off >>= 1)
      acc[j] += __shfl_down(acc[j], off, 32);
  }
  const int wave = tid >> 5, lane = tid & 31;
  if (lane == 0) {
#pragma unroll
    for (int j = 0; j < DHEAD; ++j) part[wave][j] = acc[j];
  }
  __syncthreads();
  if (tid < DHEAD) {
    float tot = 0.f;
#pragma unroll
    for (int w = 0; w < 8; ++w) tot += part[w][tid];
    rctx[b * DMODEL + h * DHEAD + tid] = tot / gsum;
  }
}

// ---------------------------------------------------------------------------
extern "C" void kernel_launch(void* const* d_in, const int* in_sizes, int n_in,
                              void* d_out, int out_size, void* d_ws,
                              size_t ws_size, hipStream_t stream) {
  const float* e = (const float*)d_in[0];
  // Param dicts may be flattened in JAX-sorted key order (k_b first, 256 elems)
  // or insertion order (q_w first, 65536 elems). Detect via in_sizes.
  int iqw, iqb, ikw, ikb, ivw, ivb, iow, iob;
  if (in_sizes[1] == DMODEL) {  // sorted: k_b,k_w,o_b,o_w,q_b,q_w,v_b,v_w
    ikb = 0; ikw = 1; iob = 2; iow = 3; iqb = 4; iqw = 5; ivb = 6; ivw = 7;
  } else {                      // insertion: q_w,q_b,k_w,k_b,v_w,v_b,o_w,o_b
    iqw = 0; iqb = 1; ikw = 2; ikb = 3; ivw = 4; ivb = 5; iow = 6; iob = 7;
  }
  const float* sat[8];
  const float* rel[8];
  for (int i = 0; i < 8; ++i) sat[i] = (const float*)d_in[1 + i];
  for (int i = 0; i < 8; ++i) rel[i] = (const float*)d_in[9 + i];
  const float* ln_sat_w = (const float*)d_in[17];
  const float* ln_sat_b = (const float*)d_in[18];
  const float* ln_rel_w = (const float*)d_in[19];
  const float* ln_rel_b = (const float*)d_in[20];

  // ---- workspace carve (~114 MB) ----
  char* wsp = (char*)d_ws;
  size_t off = 0;
  auto carve = [&](size_t bytes) -> char* {
    char* p = wsp + off;
    off += (bytes + 255) & ~(size_t)255;
    return p;
  };
  const size_t ACT_B16 = (size_t)NTOK * DMODEL * sizeof(__bf16);  // 16 MB
  __bf16* Wbf  = (__bf16*)carve(8 * 65536 * sizeof(__bf16));  // 8 weight mats
  __bf16* xbf  = (__bf16*)carve(ACT_B16);  // bf16 of current h (init: e)
  __bf16* ctxb = (__bf16*)carve(ACT_B16);  // satellite attention context
  __bf16* Qbf  = (__bf16*)carve(ACT_B16);  // Q(h); reused for O-proj output
  __bf16* Kbf  = (__bf16*)carve(ACT_B16);  // K(h); reused for relay K(h)
  __bf16* Vbf  = (__bf16*)carve(ACT_B16);  // V(h); reused for relay V(h)
  __bf16* Kebf = (__bf16*)carve(ACT_B16);  // K_sat(e), persistent
  __bf16* Vebf = (__bf16*)carve(ACT_B16);  // V_sat(e), persistent
  float*  sf   = (float*)carve(BATCH * DMODEL * sizeof(float));
  __bf16* ksb  = (__bf16*)carve(BATCH * DMODEL * sizeof(__bf16));
  __bf16* vsb  = (__bf16*)carve(BATCH * DMODEL * sizeof(__bf16));
  __bf16* k0b  = (__bf16*)carve(BATCH * DMODEL * sizeof(__bf16));
  __bf16* v0b  = (__bf16*)carve(BATCH * DMODEL * sizeof(__bf16));
  float*  qsf  = (float*)carve(BATCH * DMODEL * sizeof(float));
  float*  rctx = (float*)carve(BATCH * DMODEL * sizeof(float));
  (void)ws_size;

  float* hout = (float*)d_out;                       // (8,4096,256)
  float* sout = hout + (size_t)NTOK * DMODEL;        // (8,256)

  // bf16 weights: slots 0..7 = sat{q,k,v,o}, rel{q,k,v,o}
  const float* wsrc[8] = {sat[iqw], sat[ikw], sat[ivw], sat[iow],
                          rel[iqw], rel[ikw], rel[ivw], rel[iob == 7 ? iow : iow]};
  wsrc[7] = rel[iow];
  for (int i = 0; i < 8; ++i)
    cast_f2b_kernel<<<64, 256, 0, stream>>>(wsrc[i], Wbf + (size_t)i * 65536,
                                            65536);
  // bf16 of e (initial h)
  cast_f2b_kernel<<<(NTOK * DMODEL) / 1024, 256, 0, stream>>>(
      e, xbf, NTOK * DMODEL);
  // s0 = mean over seq
  mean_seq_kernel<<<BATCH, 256, 0, stream>>>(e, sf);

  const size_t WLDS = (size_t)DMODEL * DMODEL * sizeof(__bf16);  // 128 KB
#define GEMM(Xp, wi, bp, outbf)                                        \
  gemm_xwt_kernel<<<NTOK / 128, 256, WLDS, stream>>>(                  \
      (Xp), Wbf + (size_t)(wi) * 65536, (bp), nullptr, (outbf))

  // one-time K_sat(e), V_sat(e)  (xbf currently holds e)
  GEMM(xbf, 1, sat[ikb], Kebf);
  GEMM(xbf, 2, sat[ivb], Vebf);

  for (int cyc = 0; cyc < 2; ++cyc) {
    // ---- satellite ----
    GEMM(xbf, 0, sat[iqb], Qbf);  // Q(h)
    GEMM(xbf, 1, sat[ikb], Kbf);  // K(h)
    GEMM(xbf, 2, sat[ivb], Vbf);  // V(h)
    matvec256_kernel<<<BATCH, 256, 0, stream>>>(
        sf, sat[ikw], sat[ikb], nullptr, ksb, nullptr, nullptr, nullptr, 0);
    matvec256_kernel<<<BATCH, 256, 0, stream>>>(
        sf, sat[ivw], sat[ivb], nullptr, vsb, nullptr, nullptr, nullptr, 0);
    sat_attn_kernel<<<(NTOK * NHEADS) / 256, 256, 0, stream>>>(
        Qbf, Kbf, Vbf, Kebf, Vebf, ksb, vsb, ctxb);
    GEMM(ctxb, 3, sat[iob], Qbf);  // O-proj -> Qbf (reuse)
    relu_ln_rows_kernel<<<NTOK, 256, 0, stream>>>(Qbf, ln_sat_w, ln_sat_b,
                                                  hout, xbf);  // new h
    // ---- relay ----
    matvec256_kernel<<<BATCH, 256, 0, stream>>>(
        sf, rel[iqw], rel[iqb], qsf, nullptr, nullptr, nullptr, nullptr, 0);
    matvec256_kernel<<<BATCH, 256, 0, stream>>>(
        sf, rel[ikw], rel[ikb], nullptr, k0b, nullptr, nullptr, nullptr, 0);
    matvec256_kernel<<<BATCH, 256, 0, stream>>>(
        sf, rel[ivw], rel[ivb], nullptr, v0b, nullptr, nullptr, nullptr, 0);
    GEMM(xbf, 5, rel[ikb], Kbf);  // K_rel(h)
    GEMM(xbf, 6, rel[ivb], Vbf);  // V_rel(h)
    rel_attn_kernel<<<BATCH * NHEADS, 256, 0, stream>>>(qsf, Kbf, Vbf, k0b,
                                                        v0b, rctx);
    matvec256_kernel<<<BATCH, 256, 0, stream>>>(
        rctx, rel[iow], rel[iob], sf, nullptr, sout, ln_rel_w, ln_rel_b, 1);
  }
#undef GEMM
}
